// _RPN_48060684042854
// MI455X (gfx1250) — matrix operand order repaired and checked by reading, not compiled, not run on hardware
//
#include <hip/hip_runtime.h>

// ---------- types ----------
typedef __attribute__((ext_vector_type(16))) _Float16 v16h;
typedef __attribute__((ext_vector_type(8)))  _Float16 v8h;
typedef __attribute__((ext_vector_type(8)))  float    v8f;

// ---------- problem constants ----------
#define BB   4
#define CC   512
#define HH   75
#define WW   100
#define HP   77                 // padded height (1 halo each side)
#define WP   114                // padded width  (covers iw in [-1,112])
#define XW   112                // x16 padded width (7*16)
#define NA   9
#define NPOS (HH*WW)            // 7500
#define NANC (NPOS*NA)          // 67500
#define WT   7                  // ceil(100/16)
#define HB   19                 // ceil(75/4) h-blocks of 4 rows
#define NTH  (BB*HB*WT)         // 532 conv super-tiles
#define NT   (BB*HH*WT)         // 2100 head tiles
#define KSTEPS_CONV 144         // 9 * (512/32)
#define KSTEPS_HEAD 16          // 512/32
#define PRE  6000
#define POST 300
#define SELN 8192
#define SROW 520                // LDS row stride in f16 (512 + 8 pad) -> bank-tiled

// canonical py-faster-rcnn anchors (base 16, ratios .5/1/2, scales 8/16/32)
__constant__ float ANCH[9][4] = {
    {-84.f,  -40.f,  99.f,  55.f}, {-176.f,  -88.f, 191.f, 103.f}, {-360.f, -184.f, 375.f, 199.f},
    {-56.f,  -56.f,  71.f,  71.f}, {-120.f, -120.f, 135.f, 135.f}, {-248.f, -248.f, 263.f, 263.f},
    {-36.f,  -80.f,  51.f,  95.f}, { -80.f, -168.f,  95.f, 183.f}, {-168.f, -344.f, 183.f, 359.f}};

// ---------- 1) f32 NCHW -> zero-padded f16 NHWC (77 x 114 halo) ----------
__global__ void convert_in_kernel(const float* __restrict__ in, _Float16* __restrict__ out) {
    int tid = blockIdx.x * 256 + threadIdx.x;
    if (tid >= BB * HP * WP * CC) return;
    int c = tid & 511;
    int rest = tid >> 9;
    int wp = rest % WP; rest /= WP;
    int hp = rest % HP; int b = rest / HP;
    _Float16 v = (_Float16)0.f;
    if (hp >= 1 && hp <= HH && wp >= 1 && wp <= WW)
        v = (_Float16)in[(((size_t)b * CC + c) * HH + (hp - 1)) * WW + (wp - 1)];
    out[tid] = v;
}

// ---------- 2) pack conv weights into WMMA A-fragment lane layout ----------
// wpack[((mtile*144 + ks)*32 + lane)*16 + j], K order: k = kpos*512 + c
__global__ void pack_convw_kernel(const float* __restrict__ Wc, _Float16* __restrict__ wpack) {
    int ks = blockIdx.x;          // 0..143
    int mtile = blockIdx.y;       // 0..31
    int lane = threadIdx.x;       // 0..31
    int half = lane >> 4, mr = lane & 15;
    int m = mtile * 16 + mr;
    int p = ks / 16, cb = (ks % 16) * 32;
    int kh = p / 3, kw = p % 3;
    _Float16* dst = wpack + (((size_t)mtile * KSTEPS_CONV + ks) * 32 + lane) * 16;
    for (int j = 0; j < 16; ++j) {
        // A K order used here (contiguous, B matches): lanes<16: K 0..15 ; lanes>=16: K 16..31
        int kl = half * 16 + j;
        int c = cb + kl;
        dst[j] = (_Float16)Wc[(((size_t)m * CC + c) * 3 + kh) * 3 + kw];
    }
}

// ---------- 3) pack head weights (cls18 + bbox36 + ac9 + 1 pad = 64 rows) ----------
__global__ void pack_headw_kernel(const float* __restrict__ Wcls, const float* __restrict__ Wbbox,
                                  const float* __restrict__ Wac, const float* __restrict__ bcls,
                                  const float* __restrict__ bbbox, const float* __restrict__ bac,
                                  _Float16* __restrict__ hpack, float* __restrict__ hbias) {
    int ks = blockIdx.x;          // 0..15
    int mtile = blockIdx.y;       // 0..3
    int lane = threadIdx.x;       // 0..31
    int half = lane >> 4, mr = lane & 15;
    int m = mtile * 16 + mr;
    int cb = ks * 32;
    _Float16* dst = hpack + (((size_t)mtile * KSTEPS_HEAD + ks) * 32 + lane) * 16;
    for (int j = 0; j < 16; ++j) {
        int kl = half * 16 + j;
        int c = cb + kl;
        float v = (m < 18) ? Wcls[m * CC + c]
                : (m < 54) ? Wbbox[(m - 18) * CC + c]
                : (m < 63) ? Wac[(m - 54) * CC + c] : 0.f;
        dst[j] = (_Float16)v;
    }
    if (ks == 0 && mtile == 0) {
        for (int t = lane; t < 64; t += 32)
            hbias[t] = (t < 18) ? bcls[t] : (t < 54) ? bbbox[t - 18] : (t < 63) ? bac[t - 54] : 0.f;
    }
}

// ---------- 4) 3x3 conv as implicit GEMM ----------
// Each wave: 32 out-channels x 64 positions (2 M-tiles x 4 rows x 16 cols): one K-step
// uses 2 A-fragments + 4 LDS B-fragments for 8 WMMAs (halves LDS + A traffic per WMMA).
// B staged per kernel-position p via global_load_async_to_lds_b128, double-buffered
// (2 x 66,560 B LDS), ASYNCcnt-synced. 8 waves cover 16 M-tiles -> grid.y = 2.
__global__ __launch_bounds__(256) void conv_gemm_kernel(const _Float16* __restrict__ in16p,
                                                        const _Float16* __restrict__ wpack,
                                                        const float* __restrict__ bconv,
                                                        _Float16* __restrict__ x16p) {
    __shared__ _Float16 sb[2 * 4 * 16 * SROW];   // 133,120 B (of 320 KB)
    const int nt = blockIdx.x;                   // 0..531
    const int wt = nt % WT;
    int t = nt / WT;
    const int hb = t % HB;
    const int b = t / HB;
    const int h0 = hb * 4;                       // first of 4 rows
    const int tid = threadIdx.x;
    const int wave = tid >> 5;                   // 0..7
    const int lane = tid & 31;
    const int mtile0 = blockIdx.y * 16 + wave * 2;  // pair of M-tiles (0..31)
    const int half = lane >> 4;
    const int nl = lane & 15;
    const int w = wt * 16 + nl;                  // output column (0..111, stored padded)

    const uint32_t lds_base = (uint32_t)(uintptr_t)(&sb[0]);

    // cooperative async copy: 4 contiguous 16KB row-tiles (16 positions x 512 ch, f16)
    auto issue_copy = [&](int p, int bufsel) {
        const int dh = p / 3 - 1, dw = p % 3 - 1;
        const int iw1 = wt * 16 + dw + 1;                    // padded col of pos 0
        const uint32_t dbase = lds_base + (uint32_t)bufsel * (4 * 16 * SROW * 2);
#pragma unroll
        for (int r = 0; r < 4; ++r) {
            int ih1 = h0 + r + dh + 1;                       // padded row
            ih1 = ih1 > (HP - 1) ? (HP - 1) : ih1;           // clamp (garbage rows unused)
            const _Float16* src = in16p + (((size_t)b * HP + ih1) * WP + iw1) * CC;
            const uint32_t rbase = dbase + (uint32_t)r * (16 * SROW * 2);
#pragma unroll
            for (int q = 0; q < 4; ++q) {
                int ch = q * 256 + tid;                      // 0..1023 chunks of 16B
                int pos = ch >> 6, cc = ch & 63;
                uint64_t ga = (uint64_t)(uintptr_t)(src + ch * 8);
                uint32_t la = rbase + (uint32_t)(pos * (SROW * 2) + cc * 16);
                asm volatile("global_load_async_to_lds_b128 %0, %1, off"
                             :: "v"(la), "v"(ga) : "memory");
            }
        }
    };

    v8f acc[2][4] = {};
    const _Float16* Abase0 = wpack + (((size_t)(mtile0 + 0) * KSTEPS_CONV) * 32 + lane) * 16;
    const _Float16* Abase1 = wpack + (((size_t)(mtile0 + 1) * KSTEPS_CONV) * 32 + lane) * 16;

    issue_copy(0, 0);
    for (int p = 0; p < 9; ++p) {
        const int cur = p & 1;
        if (p < 8) {
            issue_copy(p + 1, cur ^ 1);
            asm volatile("s_wait_asynccnt 0x10" ::: "memory");  // copy for p done (in-order)
        } else {
            asm volatile("s_wait_asynccnt 0x0" ::: "memory");
        }
        __syncthreads();                                        // all waves' tiles visible
        const _Float16* sbp = sb + cur * (4 * 16 * SROW) + nl * SROW + half * 16;
#pragma unroll 2
        for (int cs = 0; cs < 16; ++cs) {
            v16h a0 = *(const v16h*)(Abase0 + (size_t)(p * 16 + cs) * 512);
            v16h a1 = *(const v16h*)(Abase1 + (size_t)(p * 16 + cs) * 512);
            v16h b0 = *(const v16h*)(sbp + 0 * (16 * SROW) + cs * 32);
            v16h b1 = *(const v16h*)(sbp + 1 * (16 * SROW) + cs * 32);
            v16h b2 = *(const v16h*)(sbp + 2 * (16 * SROW) + cs * 32);
            v16h b3 = *(const v16h*)(sbp + 3 * (16 * SROW) + cs * 32);
            acc[0][0] = __builtin_amdgcn_wmma_f32_16x16x32_f16(false, a0, false, b0, (short)0, acc[0][0], false, false);
            acc[1][0] = __builtin_amdgcn_wmma_f32_16x16x32_f16(false, a1, false, b0, (short)0, acc[1][0], false, false);
            acc[0][1] = __builtin_amdgcn_wmma_f32_16x16x32_f16(false, a0, false, b1, (short)0, acc[0][1], false, false);
            acc[1][1] = __builtin_amdgcn_wmma_f32_16x16x32_f16(false, a1, false, b1, (short)0, acc[1][1], false, false);
            acc[0][2] = __builtin_amdgcn_wmma_f32_16x16x32_f16(false, a0, false, b2, (short)0, acc[0][2], false, false);
            acc[1][2] = __builtin_amdgcn_wmma_f32_16x16x32_f16(false, a1, false, b2, (short)0, acc[1][2], false, false);
            acc[0][3] = __builtin_amdgcn_wmma_f32_16x16x32_f16(false, a0, false, b3, (short)0, acc[0][3], false, false);
            acc[1][3] = __builtin_amdgcn_wmma_f32_16x16x32_f16(false, a1, false, b3, (short)0, acc[1][3], false, false);
        }
        __syncthreads();                                        // done reading before reuse
    }

#pragma unroll
    for (int mi = 0; mi < 2; ++mi) {
        const int m0 = (mtile0 + mi) * 16 + half * 8;  // C/D: vgpr i -> row m0+i, col = lane&15
#pragma unroll
        for (int r = 0; r < 4; ++r) {
            if (h0 + r < HH) {
                v8h o;
#pragma unroll
                for (int i = 0; i < 8; ++i) {
                    float v = acc[mi][r][i] + bconv[m0 + i];
                    o[i] = (_Float16)(v > 0.f ? v : 0.f);
                }
                *(v8h*)(x16p + (((size_t)b * HH + (h0 + r)) * XW + w) * CC + m0) = o;
            }
        }
    }
}

// ---------- 5) 1x1 heads as one M=64 WMMA GEMM ----------
__global__ __launch_bounds__(128) void head_gemm_kernel(const _Float16* __restrict__ x16p,
                                                        const _Float16* __restrict__ hpack,
                                                        const float* __restrict__ hbias,
                                                        float* __restrict__ raw) {
    const int nt = blockIdx.x;
    const int wt = nt % WT;
    int t = nt / WT;
    const int h = t % HH;
    const int b = t / HH;
    const int wave = threadIdx.x >> 5;         // mtile 0..3
    const int lane = threadIdx.x & 31;
    const int half = lane >> 4, nl = lane & 15;
    const int w = wt * 16 + nl;

    v8f acc = {};
    const _Float16* Abase = hpack + (((size_t)wave * KSTEPS_HEAD) * 32 + lane) * 16;
    const _Float16* bp = x16p + (((size_t)b * HH + h) * XW + w) * CC + half * 16;
#pragma unroll 4
    for (int cs = 0; cs < 16; ++cs) {
        v16h a = *(const v16h*)(Abase + (size_t)cs * 512);
        v16h bbv = *(const v16h*)(bp + cs * 32);                // unconditional (padded x16)
        acc = __builtin_amdgcn_wmma_f32_16x16x32_f16(false, a, false, bbv,
                                                     (short)0, acc, false, false);
    }
    if (w < WW) {
        const int m0 = wave * 16 + half * 8;
        v8f o;
#pragma unroll
        for (int i = 0; i < 8; ++i) o[i] = acc[i] + hbias[m0 + i];
        *(v8f*)(raw + (((size_t)b * HH + h) * WW + w) * 64 + m0) = o;
    }
}

// ---------- 6) softmax/sigmoid + anchor decode + clip ----------
__global__ void proposal_kernel(const float* __restrict__ raw, const float* __restrict__ im_info,
                                float* __restrict__ scores, float* __restrict__ boxes,
                                float* __restrict__ acf) {
    int tid = blockIdx.x * 256 + threadIdx.x;
    if (tid >= BB * NANC) return;
    int a = tid % NA;
    int pos = tid / NA;                        // b*7500 + h*100 + w
    int b = pos / NPOS;
    int hw = pos % NPOS;
    int h = hw / WW, w = hw % WW;
    const float* r = raw + (size_t)pos * 64;
    float score = 1.f / (1.f + __expf(r[a] - r[NA + a]));          // sigmoid(fg-bg)
    float d0 = r[18 + a * 4 + 0], d1 = r[18 + a * 4 + 1];
    float d2 = r[18 + a * 4 + 2], d3 = r[18 + a * 4 + 3];
    float acv = 1.f / (1.f + __expf(-r[54 + a]));
    float sx = w * 16.f, sy = h * 16.f;
    float x1a = ANCH[a][0] + sx, y1a = ANCH[a][1] + sy;
    float x2a = ANCH[a][2] + sx, y2a = ANCH[a][3] + sy;
    float aw = x2a - x1a + 1.f, ah = y2a - y1a + 1.f;
    float acx = x1a + 0.5f * aw, acy = y1a + 0.5f * ah;
    float pcx = d0 * aw + acx, pcy = d1 * ah + acy;
    float pw = __expf(d2) * aw, ph = __expf(d3) * ah;
    float imh = im_info[b * 3 + 0], imw = im_info[b * 3 + 1];
    float x1 = fminf(fmaxf(pcx - 0.5f * pw, 0.f), imw - 1.f);
    float y1 = fminf(fmaxf(pcy - 0.5f * ph, 0.f), imh - 1.f);
    float x2 = fminf(fmaxf(pcx + 0.5f * pw, 0.f), imw - 1.f);
    float y2 = fminf(fmaxf(pcy + 0.5f * ph, 0.f), imh - 1.f);
    int n = hw * NA + a;
    scores[(size_t)b * NANC + n] = score;
    float* bx = boxes + ((size_t)b * NANC + n) * 4;
    bx[0] = x1; bx[1] = y1; bx[2] = x2; bx[3] = y2;
    acf[(size_t)b * NANC + n] = acv;
}

// ---------- 7) top-k machinery ----------
__device__ __forceinline__ unsigned fkey(float f) {
    unsigned u = __float_as_uint(f);
    return (u & 0x80000000u) ? ~u : (u | 0x80000000u);
}

__global__ void init_kernel(unsigned long long* sel, unsigned* hist, unsigned* cnt) {
    int tid = blockIdx.x * 256 + threadIdx.x;
    if (tid < BB * SELN) sel[tid] = 0ull;
    if (tid < BB * 256) hist[tid] = 0u;
    if (tid < BB * 2) cnt[tid] = 0u;
}

__global__ void hist_kernel(const float* __restrict__ scores, unsigned* __restrict__ hist) {
    int tid = blockIdx.x * 256 + threadIdx.x;
    if (tid >= BB * NANC) return;
    int b = tid / NANC;
    unsigned k = fkey(scores[tid]);
    atomicAdd(&hist[b * 256 + (k >> 24)], 1u);
}

__global__ void scan_kernel(const unsigned* __restrict__ hist, int* __restrict__ tinfo) {
    if (threadIdx.x != 0) return;
    int b = blockIdx.x;
    unsigned cum = 0; int tbin = 0; unsigned above = 0;
    for (int bin = 255; bin >= 0; --bin) {
        unsigned c = hist[b * 256 + bin];
        if (cum + c >= (unsigned)PRE) { tbin = bin; above = cum; break; }
        cum += c;
    }
    tinfo[b * 2 + 0] = tbin;
    tinfo[b * 2 + 1] = (int)above;
}

__global__ void compact_kernel(const float* __restrict__ scores, const int* __restrict__ tinfo,
                               unsigned* __restrict__ cnt, unsigned long long* __restrict__ sel) {
    int tid = blockIdx.x * 256 + threadIdx.x;
    if (tid >= BB * NANC) return;
    int b = tid / NANC, n = tid % NANC;
    unsigned k = fkey(scores[tid]);
    int bin = (int)(k >> 24);
    int tbin = tinfo[b * 2 + 0];
    unsigned above = (unsigned)tinfo[b * 2 + 1];
    unsigned long long pk = ((unsigned long long)k << 32) | (unsigned)n;
    if (bin > tbin) {
        unsigned p = atomicAdd(&cnt[b * 2 + 0], 1u);
        sel[(size_t)b * SELN + p] = pk;
    } else if (bin == tbin) {
        unsigned e = atomicAdd(&cnt[b * 2 + 1], 1u);
        unsigned p = above + e;
        if (p < (unsigned)PRE) sel[(size_t)b * SELN + p] = pk;
    }
}

// ---------- 8) per-batch bitonic sort (desc) of 8192 (key,idx) pairs + gather ----------
__global__ __launch_bounds__(1024) void sort_gather_kernel(const unsigned long long* __restrict__ sel,
                                                           const float* __restrict__ scores,
                                                           const float* __restrict__ boxes,
                                                           const float* __restrict__ acf,
                                                           float* __restrict__ tb, float* __restrict__ ts,
                                                           float* __restrict__ ta) {
    __shared__ unsigned long long s[SELN];      // 64KB
    int b = blockIdx.x, tid = threadIdx.x;
    for (int i = tid; i < SELN; i += 1024) s[i] = sel[(size_t)b * SELN + i];
    __syncthreads();
    for (int k = 2; k <= SELN; k <<= 1) {
        for (int j = k >> 1; j > 0; j >>= 1) {
            for (int i = tid; i < SELN; i += 1024) {
                int ixj = i ^ j;
                if (ixj > i) {
                    unsigned long long x = s[i], y = s[ixj];
                    bool up = (i & k) == 0;                 // descending overall
                    if (up ? (x < y) : (x > y)) { s[i] = y; s[ixj] = x; }
                }
            }
            __syncthreads();
        }
    }
    for (int i = tid; i < PRE; i += 1024) {
        unsigned idx = (unsigned)(s[i] & 0xffffffffu);
        size_t src = (size_t)b * NANC + idx;
        ts[(size_t)b * PRE + i] = scores[src];
        ta[(size_t)b * PRE + i] = acf[src];
        const float* bs = boxes + src * 4;
        float* bd = tb + ((size_t)b * PRE + i) * 4;
        bd[0] = bs[0]; bd[1] = bs[1]; bd[2] = bs[2]; bd[3] = bs[3];
    }
}

// ---------- 9) greedy NMS, 300 fixed picks ----------
__global__ __launch_bounds__(1024) void nms_kernel(const float* __restrict__ tb,
                                                   const float* __restrict__ ts,
                                                   int* __restrict__ keep) {
    __shared__ float ss[PRE];
    __shared__ float ar[PRE];
    __shared__ float rv[1024];
    __shared__ int   ri[1024];
    __shared__ float pb[5];
    int b = blockIdx.x, tid = threadIdx.x;
    for (int i = tid; i < PRE; i += 1024) {
        ss[i] = ts[(size_t)b * PRE + i];
        const float* bx = tb + ((size_t)b * PRE + i) * 4;
        ar[i] = (bx[2] - bx[0] + 1.f) * (bx[3] - bx[1] + 1.f);
    }
    __syncthreads();
    for (int it = 0; it < POST; ++it) {
        float bv = -3.4e38f; int bi = 0x7fffffff;
        for (int i = tid; i < PRE; i += 1024) {
            float v = ss[i];
            if (v > bv || (v == bv && i < bi)) { bv = v; bi = i; }
        }
        rv[tid] = bv; ri[tid] = bi;
        __syncthreads();
        for (int s = 512; s > 0; s >>= 1) {
            if (tid < s) {
                float ov = rv[tid + s]; int oi = ri[tid + s];
                if (ov > rv[tid] || (ov == rv[tid] && oi < ri[tid])) { rv[tid] = ov; ri[tid] = oi; }
            }
            __syncthreads();
        }
        if (tid == 0) {
            int k = ri[0];
            keep[b * POST + it] = k;
            const float* bx = tb + ((size_t)b * PRE + k) * 4;
            pb[0] = bx[0]; pb[1] = bx[1]; pb[2] = bx[2]; pb[3] = bx[3]; pb[4] = ar[k];
        }
        __syncthreads();
        float px1 = pb[0], py1 = pb[1], px2 = pb[2], py2 = pb[3], pa = pb[4];
        for (int i = tid; i < PRE; i += 1024) {
            const float* bx = tb + ((size_t)b * PRE + i) * 4;
            float xx1 = fmaxf(px1, bx[0]), yy1 = fmaxf(py1, bx[1]);
            float xx2 = fminf(px2, bx[2]), yy2 = fminf(py2, bx[3]);
            float inter = fmaxf(xx2 - xx1 + 1.f, 0.f) * fmaxf(yy2 - yy1 + 1.f, 0.f);
            float iou = inter / (pa + ar[i] - inter);
            if (iou > 0.7f) ss[i] = -1e30f;
        }
        __syncthreads();
    }
}

// ---------- 10) final [4,300,7] gather ----------
__global__ void out_gather_kernel(const float* __restrict__ tb, const float* __restrict__ ts,
                                  const float* __restrict__ ta, const int* __restrict__ keep,
                                  float* __restrict__ out) {
    int tid = blockIdx.x * 256 + threadIdx.x;
    if (tid >= BB * POST) return;
    int b = tid / POST, i = tid % POST;
    int k = keep[b * POST + i];
    float* o = out + (size_t)tid * 7;
    const float* bx = tb + ((size_t)b * PRE + k) * 4;
    o[0] = (float)b;
    o[1] = bx[0]; o[2] = bx[1]; o[3] = bx[2]; o[4] = bx[3];
    o[5] = ts[(size_t)b * PRE + k];
    o[6] = ta[(size_t)b * PRE + k];
}

// ---------- launcher ----------
extern "C" void kernel_launch(void* const* d_in, const int* in_sizes, int n_in,
                              void* d_out, int out_size, void* d_ws, size_t ws_size,
                              hipStream_t stream) {
    (void)in_sizes; (void)n_in; (void)out_size; (void)ws_size;
    const float* base_feat = (const float*)d_in[0];
    const float* im_info   = (const float*)d_in[1];
    const float* W_conv    = (const float*)d_in[5];
    const float* b_conv    = (const float*)d_in[6];
    const float* W_cls     = (const float*)d_in[7];
    const float* b_cls     = (const float*)d_in[8];
    const float* W_bbox    = (const float*)d_in[9];
    const float* b_bbox    = (const float*)d_in[10];
    const float* W_ac      = (const float*)d_in[11];
    const float* b_ac      = (const float*)d_in[12];
    float* out = (float*)d_out;

    char* ws = (char*)d_ws;
    const size_t o_in16p  = 0;                             // 35,954,688 B (raw aliases this later)
    const size_t o_wpack  = o_in16p + 35954688;            //  4,718,592
    const size_t o_x16    = o_wpack + 4718592;             // 34,406,400
    const size_t o_hpack  = o_x16   + 34406400;            //     65,536
    const size_t o_hbias  = o_hpack + 65536;               //        256
    const size_t o_scores = o_hbias + 256;                 //  1,080,000
    const size_t o_boxes  = o_scores + 1080000;            //  4,320,000
    const size_t o_acf    = o_boxes  + 4320000;            //  1,080,000
    const size_t o_sel    = o_acf    + 1080000;            //    262,144
    const size_t o_hist   = o_sel    + 262144;             //      4,096
    const size_t o_cnt    = o_hist   + 4096;               //         64
    const size_t o_tinfo  = o_cnt    + 64;                 //         64
    const size_t o_tb     = o_tinfo  + 64;                 //    384,000
    const size_t o_ts     = o_tb     + 384000;             //     96,000
    const size_t o_ta     = o_ts     + 96000;              //     96,000
    const size_t o_keep   = o_ta     + 96000;              //      4,800

    _Float16* in16p = (_Float16*)(ws + o_in16p);
    _Float16* wpack = (_Float16*)(ws + o_wpack);
    _Float16* x16p  = (_Float16*)(ws + o_x16);
    _Float16* hpack = (_Float16*)(ws + o_hpack);
    float* hbias    = (float*)(ws + o_hbias);
    float* raw      = (float*)(ws + o_in16p);   // alias: conv input no longer needed
    float* scores   = (float*)(ws + o_scores);
    float* boxes    = (float*)(ws + o_boxes);
    float* acf      = (float*)(ws + o_acf);
    unsigned long long* sel = (unsigned long long*)(ws + o_sel);
    unsigned* hist  = (unsigned*)(ws + o_hist);
    unsigned* cnt   = (unsigned*)(ws + o_cnt);
    int* tinfo      = (int*)(ws + o_tinfo);
    float* tb       = (float*)(ws + o_tb);
    float* ts       = (float*)(ws + o_ts);
    float* ta       = (float*)(ws + o_ta);
    int* keep       = (int*)(ws + o_keep);

    convert_in_kernel<<<(BB * HP * WP * CC + 255) / 256, 256, 0, stream>>>(base_feat, in16p);
    pack_convw_kernel<<<dim3(KSTEPS_CONV, 32), 32, 0, stream>>>(W_conv, wpack);
    pack_headw_kernel<<<dim3(KSTEPS_HEAD, 4), 32, 0, stream>>>(W_cls, W_bbox, W_ac,
                                                               b_cls, b_bbox, b_ac, hpack, hbias);
    conv_gemm_kernel<<<dim3(NTH, 2), 256, 0, stream>>>(in16p, wpack, b_conv, x16p);
    head_gemm_kernel<<<NT, 128, 0, stream>>>(x16p, hpack, hbias, raw);
    proposal_kernel<<<(BB * NANC + 255) / 256, 256, 0, stream>>>(raw, im_info, scores, boxes, acf);
    init_kernel<<<(BB * SELN + 255) / 256, 256, 0, stream>>>(sel, hist, cnt);
    hist_kernel<<<(BB * NANC + 255) / 256, 256, 0, stream>>>(scores, hist);
    scan_kernel<<<BB, 32, 0, stream>>>(hist, tinfo);
    compact_kernel<<<(BB * NANC + 255) / 256, 256, 0, stream>>>(scores, tinfo, cnt, sel);
    sort_gather_kernel<<<BB, 1024, 0, stream>>>(sel, scores, boxes, acf, tb, ts, ta);
    nms_kernel<<<BB, 1024, 0, stream>>>(tb, ts, keep);
    out_gather_kernel<<<(BB * POST + 255) / 256, 256, 0, stream>>>(tb, ts, ta, keep, out);
}
